// ContrastiveLoss_83734682403106
// MI455X (gfx1250) — compile-verified, matrix operand order checked by reading, not run on hardware
//
#include <hip/hip_runtime.h>
#include <hip/hip_bf16.h>
#include <math.h>

typedef __attribute__((ext_vector_type(16))) _Float16 v16h;
typedef __attribute__((ext_vector_type(8)))  float    v8f;

#define DIM 256            // embedding dim (fixed by reference)
#define KSTEPS (DIM / 32)  // 8 WMMA K-steps of 32
#define WAVES_PER_BLOCK 4
#define CSPLIT 4           // column-range split for occupancy
#define SHIFT 2.0f         // logits bounded in [-2,2]: fixed, exact logsumexp shift
#define TILE_HALFS (16 * DIM)   // halfs per 16-row B tile
// exp(2*dot - SHIFT) == exp2(dot * C2 - C2) with C2 = 2*log2(e)
#define C2LOG2E 2.88539008177792681f

// ---------------------------------------------------------------------------
// Kernel 0: zero the scalar output
// ---------------------------------------------------------------------------
__global__ void zero_kernel(float* out) {
    if (threadIdx.x == 0) out[0] = 0.0f;
}

// ---------------------------------------------------------------------------
// Kernel 1: L2-normalize each row of [emb_i; emb_j] and store as f16.
// ---------------------------------------------------------------------------
__global__ void normalize_kernel(const float* __restrict__ emb_i,
                                 const float* __restrict__ emb_j,
                                 _Float16* __restrict__ reps, int N) {
    const int r = blockIdx.x;
    const int t = threadIdx.x;
    const float* src = (r < N) ? (emb_i + (size_t)r * DIM)
                               : (emb_j + (size_t)(r - N) * DIM);
    float x  = src[t];
    float ss = x * x;
    #pragma unroll
    for (int off = 16; off >= 1; off >>= 1)
        ss += __shfl_xor(ss, off, 32);
    __shared__ float wsum[DIM / 32];
    if ((t & 31) == 0) wsum[t >> 5] = ss;
    __syncthreads();
    float tot = 0.0f;
    #pragma unroll
    for (int w = 0; w < DIM / 32; ++w) tot += wsum[w];
    const float inv = 1.0f / fmaxf(sqrtf(tot), 1e-12f);
    reps[(size_t)r * DIM + t] = (_Float16)(x * inv);
}

// ---------------------------------------------------------------------------
// Kernel 2: fused reps @ reps^T (WMMA f16 -> f32) + shifted sum-of-exp.
// Hot loop is fully branch-free; diagonal / positive tiles are recomputed
// once after the loop (bit-identical WMMA), so corrections are exact.
// ---------------------------------------------------------------------------
__global__ void __launch_bounds__(WAVES_PER_BLOCK * 32)
ntxent_wmma_kernel(const _Float16* __restrict__ reps,
                   float* __restrict__ s_part,
                   float* __restrict__ p_part,
                   int twoN, int N) {
    const int lane    = threadIdx.x & 31;
    const int wave    = threadIdx.x >> 5;
    const int l16     = lane & 15;
    const int halfSel = lane >> 4;                      // 0: K 0-15 | 1: K 16-31
    const int r0      = (blockIdx.x * WAVES_PER_BLOCK + wave) * 16;
    const int chunk   = blockIdx.y;
    const int jTiles  = twoN / 16;
    const int jPer    = jTiles / CSPLIT;
    const int jBeg    = chunk * jPer;
    const int jEnd    = jBeg + jPer;

    // Tiles (uniform per wave) where the diagonal / positive column land.
    const int jDiag = r0 >> 4;
    int pr0 = r0 + N; if (pr0 >= twoN) pr0 -= twoN;
    const int jPos  = pr0 >> 4;

    // ---- Preload all A fragments for this 16-row tile (reused jPer times) --
    v16h a[KSTEPS];
    {
        const _Float16* aBase = reps + (size_t)(r0 + l16) * DIM + halfSel * 16;
        #pragma unroll
        for (int k = 0; k < KSTEPS; ++k)
            a[k] = *(const v16h*)(aBase + k * 32);
    }

    // Per-lane state for 8 rows (row = r0 + v + 8*halfSel, cols ≡ l16 mod 16)
    float s[8], p[8];
    #pragma unroll
    for (int v = 0; v < 8; ++v) { s[v] = 0.0f; p[v] = 0.0f; }

    // ---- Preload B fragments for first column tile; running pointer --------
    const _Float16* bLoad =
        reps + (size_t)(jBeg * 16 + l16) * DIM + halfSel * 16;
    v16h b[KSTEPS];
    #pragma unroll
    for (int k = 0; k < KSTEPS; ++k)
        b[k] = *(const v16h*)(bLoad + k * 32);
    bLoad += TILE_HALFS;                     // next tile to fetch

    // =================== branch-free hot loop ==============================
    for (int j = jBeg; j < jEnd; ++j) {
        // WGP-scope prefetch 8 tiles ahead (fills all cache levels).
        {
            const _Float16* pf = bLoad + 7 * (size_t)TILE_HALFS;
            asm volatile("global_prefetch_b8 %0, off" :: "v"(pf));
        }

        // 8 back-to-back WMMAs consuming the resident b[] buffer.
        v8f acc = {};
        #pragma unroll
        for (int k = 0; k < KSTEPS; ++k)
            acc = __builtin_amdgcn_wmma_f32_16x16x32_f16(
                false, a[k], false, b[k], (short)0, acc, false, false);

        // Unconditionally issue next tile's loads (one-past-end lands in the
        // partials region of the workspace: valid memory, never consumed).
        #pragma unroll
        for (int k = 0; k < KSTEPS; ++k)
            b[k] = *(const v16h*)(bLoad + k * 32);
        bLoad += TILE_HALFS;

        // Epilogue: s += exp2(dot*C2 - C2) == exp(2*dot - SHIFT).
        // Lowers to pk_fma + v_exp + pk_add (no log2e pk_mul).
        #pragma unroll
        for (int v = 0; v < 8; ++v)
            s[v] += __builtin_amdgcn_exp2f(fmaf(acc[v], C2LOG2E, -C2LOG2E));
    }

    // =================== post-loop exact corrections =======================
    // Recompute a special tile (bit-identical to its in-loop computation).
    if (jDiag >= jBeg && jDiag < jEnd) {     // subtract the self term
        const _Float16* base =
            reps + (size_t)(jDiag * 16 + l16) * DIM + halfSel * 16;
        v16h bb[KSTEPS];
        #pragma unroll
        for (int k = 0; k < KSTEPS; ++k)
            bb[k] = *(const v16h*)(base + k * 32);
        v8f accD = {};
        #pragma unroll
        for (int k = 0; k < KSTEPS; ++k)
            accD = __builtin_amdgcn_wmma_f32_16x16x32_f16(
                false, a[k], false, bb[k], (short)0, accD, false, false);
        #pragma unroll
        for (int v = 0; v < 8; ++v)
            if (l16 == v + 8 * halfSel)      // col == row
                s[v] -= __builtin_amdgcn_exp2f(
                            fmaf(accD[v], C2LOG2E, -C2LOG2E));
    }
    if (jPos >= jBeg && jPos < jEnd) {       // capture positive-pair logit
        const _Float16* base =
            reps + (size_t)(jPos * 16 + l16) * DIM + halfSel * 16;
        v16h bb[KSTEPS];
        #pragma unroll
        for (int k = 0; k < KSTEPS; ++k)
            bb[k] = *(const v16h*)(base + k * 32);
        v8f accP = {};
        #pragma unroll
        for (int k = 0; k < KSTEPS; ++k)
            accP = __builtin_amdgcn_wmma_f32_16x16x32_f16(
                false, a[k], false, bb[k], (short)0, accP, false, false);
        #pragma unroll
        for (int v = 0; v < 8; ++v) {
            const int row = r0 + v + 8 * halfSel;
            int pc = row + N; if (pc >= twoN) pc -= twoN;
            if ((pc & 15) == l16) p[v] = accP[v] * 2.0f;
        }
    }

    // ---- Sum partials across the 16 column-residue lanes per row ----------
    #pragma unroll
    for (int v = 0; v < 8; ++v) {
        float sv = s[v], pv = p[v];
        #pragma unroll
        for (int off = 1; off <= 8; off <<= 1) {
            sv += __shfl_xor(sv, off, 32);
            pv += __shfl_xor(pv, off, 32);
        }
        if (l16 == 0) {
            const int row = r0 + v + 8 * halfSel;
            const size_t idx = (size_t)chunk * twoN + row;
            s_part[idx] = sv;
            p_part[idx] = pv;
        }
    }
}

// ---------------------------------------------------------------------------
// Kernel 3: merge CSPLIT partials per row, accumulate mean loss.
// ---------------------------------------------------------------------------
__global__ void finalize_kernel(const float* __restrict__ s_part,
                                const float* __restrict__ p_part,
                                float* __restrict__ out, int twoN) {
    const int r = blockIdx.x * blockDim.x + threadIdx.x;
    if (r >= twoN) return;
    float s = 0.0f, p = 0.0f;
    #pragma unroll
    for (int c = 0; c < CSPLIT; ++c) {
        const size_t idx = (size_t)c * twoN + r;
        s += s_part[idx];
        p += p_part[idx];
    }
    const float contrib = (SHIFT + __logf(s) - p) / (float)twoN;
    atomicAdd(out, contrib);
}

// ---------------------------------------------------------------------------
extern "C" void kernel_launch(void* const* d_in, const int* in_sizes, int n_in,
                              void* d_out, int out_size, void* d_ws, size_t ws_size,
                              hipStream_t stream) {
    const float* emb_i = (const float*)d_in[0];
    const float* emb_j = (const float*)d_in[1];
    float* out = (float*)d_out;

    const int N    = in_sizes[0] / DIM;   // 4096
    const int twoN = 2 * N;               // 8192

    // Workspace layout: [reps f16 twoN*DIM][s CSPLIT*twoN][p CSPLIT*twoN]
    char* ws = (char*)d_ws;
    _Float16* reps = (_Float16*)ws;
    size_t off = (size_t)twoN * DIM * sizeof(_Float16);
    float* s_part = (float*)(ws + off); off += (size_t)CSPLIT * twoN * sizeof(float);
    float* p_part = (float*)(ws + off);

    zero_kernel<<<1, 32, 0, stream>>>(out);

    normalize_kernel<<<twoN, DIM, 0, stream>>>(emb_i, emb_j, reps, N);

    dim3 grid(twoN / (16 * WAVES_PER_BLOCK), CSPLIT);   // (128, 4)
    ntxent_wmma_kernel<<<grid, WAVES_PER_BLOCK * 32, 0, stream>>>(
        reps, s_part, p_part, twoN, N);

    finalize_kernel<<<(twoN + 255) / 256, 256, 0, stream>>>(
        s_part, p_part, out, twoN);
}
